// PointNet2DecoderWithoutClsHead_35880156791483
// MI455X (gfx1250) — matrix operand order repaired and checked by reading, not compiled, not run on hardware
//
#include <hip/hip_runtime.h>
#include <hip/hip_bf16.h>

typedef __attribute__((ext_vector_type(16))) _Float16 v16h;
typedef __attribute__((ext_vector_type(8)))  _Float16 v8h;
typedef __attribute__((ext_vector_type(8)))  float    v8f;

#define DIST_EPS 1e-8f
#define BN_EPS   1e-5f

// ---------------------------------------------------------------------------
// f32 -> f16 conversion (weights)
// ---------------------------------------------------------------------------
__global__ void f32_to_f16_kernel(const float* __restrict__ src,
                                  _Float16* __restrict__ dst, int n) {
    int i = blockIdx.x * blockDim.x + threadIdx.x;
    if (i < n) dst[i] = (_Float16)src[i];
}

// ---------------------------------------------------------------------------
// Copy skip features points1 [B,C1,N] (f32) into X [B*N, Ktot] (f16),
// channels 0..C1.  c is the innermost thread index -> coalesced stores.
// ---------------------------------------------------------------------------
__global__ void copy_p1_kernel(const float* __restrict__ p1,
                               _Float16* __restrict__ X,
                               int B, int C1, int N, int Ktot) {
    int idx = blockIdx.x * blockDim.x + threadIdx.x;
    int tot = B * N * C1;
    if (idx >= tot) return;
    int c = idx % C1;
    int n = (idx / C1) % N;
    int b = idx / (C1 * N);
    X[(size_t)(b * N + n) * Ktot + c] = (_Float16)p1[((size_t)b * C1 + c) * N + n];
}

// ---------------------------------------------------------------------------
// 3-NN inverse-distance interpolation.
// xyz1: [B,3,N] f32, xyz2: [B,3,S] f32.
// p2 either f32 [B,C2,S] (first stage) or f16 [B*S, C2] (our stage buffers).
// Writes X[(b*N+n)*Ktot + c_off + c] in f16 (contiguous per thread).
// ---------------------------------------------------------------------------
__global__ void interp_kernel(const float* __restrict__ xyz1,
                              const float* __restrict__ xyz2,
                              const float* __restrict__ p2f,
                              const _Float16* __restrict__ p2h,
                              _Float16* __restrict__ X,
                              int B, int N, int S, int C2, int c_off, int Ktot) {
    int idx = blockIdx.x * blockDim.x + threadIdx.x;
    if (idx >= B * N) return;
    int b = idx / N;
    int n = idx % N;

    const float* x1 = xyz1 + (size_t)b * 3 * N;
    float px = x1[n], py = x1[N + n], pz = x1[2 * N + n];
    const float* x2 = xyz2 + (size_t)b * 3 * S;

    float d0 = 1e30f, d1 = 1e30f, d2 = 1e30f;
    int   i0 = 0,     i1 = 0,     i2 = 0;
    for (int s = 0; s < S; ++s) {
        float dx = px - x2[s];
        float dy = py - x2[S + s];
        float dz = pz - x2[2 * S + s];
        float d = dx * dx + dy * dy + dz * dz;
        if (d < d0)      { d2 = d1; i2 = i1; d1 = d0; i1 = i0; d0 = d; i0 = s; }
        else if (d < d1) { d2 = d1; i2 = i1; d1 = d;  i1 = s; }
        else if (d < d2) { d2 = d;  i2 = s; }
    }
    float w0 = 1.f / (d0 + DIST_EPS);
    float w1 = 1.f / (d1 + DIST_EPS);
    float w2 = 1.f / (d2 + DIST_EPS);
    float ws = w0 + w1 + w2;
    w0 /= ws; w1 /= ws; w2 /= ws;

    _Float16* xdst = X + (size_t)(b * N + n) * Ktot + c_off;
    if (p2h) {
        const _Float16* r0 = p2h + (size_t)(b * S + i0) * C2;
        const _Float16* r1 = p2h + (size_t)(b * S + i1) * C2;
        const _Float16* r2 = p2h + (size_t)(b * S + i2) * C2;
        for (int c = 0; c < C2; ++c) {
            float v = w0 * (float)r0[c] + w1 * (float)r1[c] + w2 * (float)r2[c];
            xdst[c] = (_Float16)v;
        }
    } else {
        const float* pb = p2f + (size_t)b * C2 * S;
        for (int c = 0; c < C2; ++c) {
            const float* row = pb + (size_t)c * S;
            float v = w0 * row[i0] + w1 * row[i1] + w2 * row[i2];
            xdst[c] = (_Float16)v;
        }
    }
}

// ---------------------------------------------------------------------------
// GEMM via v_wmma_f32_16x16x32_f16, wide tiles.
// Y[m, o] = sum_k Wh[o,k] * Xh[m,k] + bias[o]
// Wh: [outC, K] f16 row-major (A, K contiguous);
// Xh: [M, K]    f16 row-major (B, K contiguous);
// Y : [M, outC] f32.
// One wave computes a 32(o) x 64(m) tile: 2 A frags x 4 B frags, 8 accums.
// Fragment loads are contiguous 16B/32B vector loads:
//   A lane L (row = base+L%16, hs=L/16): halves 0..7  <- K = k0+hs*16+{0..7}
//                                        halves 8..15 <- K = k0+hs*16+8+{0..7}
//   B lane L (col = base+L%16, hs=L/16): halves 0..15 <- K = k0+hs*16+{0..15}
//   D element r: row = r + 8*hs, col = L%16  -> per-lane 8 contiguous floats.
// #pragma unroll 1 + launch_bounds(256,1): keep one fragment set live,
// no scratch spills of the accumulator tiles.
// ---------------------------------------------------------------------------
__global__ __launch_bounds__(256, 1)
void gemm_wmma_kernel(const _Float16* __restrict__ Wh,
                      const _Float16* __restrict__ Xh,
                      const float* __restrict__ bias,
                      float* __restrict__ Y,
                      int K, int M, int outC) {
    int n_mt  = M >> 6;                               // tiles of 64 along m
    int tile  = blockIdx.x * 8 + (threadIdx.x >> 5);  // 8 waves per block
    int mt    = tile % n_mt;
    int ot    = tile / n_mt;                          // tiles of 32 along o

    int lane = threadIdx.x & 31;
    int hs   = lane >> 4;
    int l16  = lane & 15;

    // Persistent row pointers, advanced by 32 halves (64 B) per K-step.
    const _Float16* ap0 = Wh + (size_t)(ot * 32 + l16) * K + hs * 16;
    const _Float16* ap1 = ap0 + (size_t)16 * K;
    const _Float16* bp0 = Xh + (size_t)(mt * 64 + 0 * 16 + l16) * K + hs * 16;
    const _Float16* bp1 = Xh + (size_t)(mt * 64 + 1 * 16 + l16) * K + hs * 16;
    const _Float16* bp2 = Xh + (size_t)(mt * 64 + 2 * 16 + l16) * K + hs * 16;
    const _Float16* bp3 = Xh + (size_t)(mt * 64 + 3 * 16 + l16) * K + hs * 16;

    v8f acc[2][4] = {};
#pragma unroll 1
    for (int k0 = 0; k0 < K; k0 += 32) {
        v8h a0lo = *(const v8h*)(ap0);
        v8h a0hi = *(const v8h*)(ap0 + 8);    // constant +16 B offset
        v8h a1lo = *(const v8h*)(ap1);
        v8h a1hi = *(const v8h*)(ap1 + 8);
        v16h af0 = __builtin_shufflevector(a0lo, a0hi, 0, 1, 2, 3, 4, 5, 6, 7,
                                           8, 9, 10, 11, 12, 13, 14, 15);
        v16h af1 = __builtin_shufflevector(a1lo, a1hi, 0, 1, 2, 3, 4, 5, 6, 7,
                                           8, 9, 10, 11, 12, 13, 14, 15);
        v16h bf0 = *(const v16h*)(bp0);       // 32 B -> b128 x2 (offset:16)
        v16h bf1 = *(const v16h*)(bp1);
        v16h bf2 = *(const v16h*)(bp2);
        v16h bf3 = *(const v16h*)(bp3);

        acc[0][0] = __builtin_amdgcn_wmma_f32_16x16x32_f16(false, af0, false, bf0, (short)0, acc[0][0], false, false);
        acc[0][1] = __builtin_amdgcn_wmma_f32_16x16x32_f16(false, af0, false, bf1, (short)0, acc[0][1], false, false);
        acc[0][2] = __builtin_amdgcn_wmma_f32_16x16x32_f16(false, af0, false, bf2, (short)0, acc[0][2], false, false);
        acc[0][3] = __builtin_amdgcn_wmma_f32_16x16x32_f16(false, af0, false, bf3, (short)0, acc[0][3], false, false);
        acc[1][0] = __builtin_amdgcn_wmma_f32_16x16x32_f16(false, af1, false, bf0, (short)0, acc[1][0], false, false);
        acc[1][1] = __builtin_amdgcn_wmma_f32_16x16x32_f16(false, af1, false, bf1, (short)0, acc[1][1], false, false);
        acc[1][2] = __builtin_amdgcn_wmma_f32_16x16x32_f16(false, af1, false, bf2, (short)0, acc[1][2], false, false);
        acc[1][3] = __builtin_amdgcn_wmma_f32_16x16x32_f16(false, af1, false, bf3, (short)0, acc[1][3], false, false);

        ap0 += 32; ap1 += 32;
        bp0 += 32; bp1 += 32; bp2 += 32; bp3 += 32;
    }
#pragma unroll
    for (int i = 0; i < 2; ++i) {
        int orow0 = ot * 32 + i * 16 + hs * 8;
#pragma unroll
        for (int j = 0; j < 4; ++j) {
            int col = mt * 64 + j * 16 + l16;
            v8f out;
#pragma unroll
            for (int r = 0; r < 8; ++r)
                out[r] = acc[i][j][r] + bias[orow0 + r];
            *(v8f*)(Y + (size_t)col * outC + orow0) = out;
        }
    }
}

// ---------------------------------------------------------------------------
// Per-channel mean / rstd over M elements (train-mode BatchNorm statistics).
// One block (256 threads) per channel.  Y layout [M, outC].
// ---------------------------------------------------------------------------
__global__ void stats_kernel(const float* __restrict__ Y,
                             float2* __restrict__ stats, int M, int outC) {
    int o = blockIdx.x;
    __shared__ float ssum[256];
    __shared__ float ssq[256];
    float s = 0.f, q = 0.f;
    for (int m = threadIdx.x; m < M; m += 256) {
        float v = Y[(size_t)m * outC + o];
        s += v;
        q += v * v;
    }
    ssum[threadIdx.x] = s;
    ssq[threadIdx.x]  = q;
    __syncthreads();
    for (int st = 128; st > 0; st >>= 1) {
        if ((int)threadIdx.x < st) {
            ssum[threadIdx.x] += ssum[threadIdx.x + st];
            ssq[threadIdx.x]  += ssq[threadIdx.x + st];
        }
        __syncthreads();
    }
    if (threadIdx.x == 0) {
        float mu  = ssum[0] / (float)M;
        float var = ssq[0] / (float)M - mu * mu;
        stats[o] = make_float2(mu, rsqrtf(var + BN_EPS));
    }
}

// ---------------------------------------------------------------------------
// BN (train) + ReLU.
// f16 path: dsth [M, outC] (coalesced read+write, same index).
// f32 path: dstf [B, outC, N] final output (n-inner mapping, coalesced write).
// ---------------------------------------------------------------------------
__global__ void bnrelu_kernel(const float* __restrict__ Y,
                              const float* __restrict__ gamma,
                              const float* __restrict__ beta,
                              const float2* __restrict__ stats,
                              _Float16* __restrict__ dsth,
                              float* __restrict__ dstf,
                              int outC, int M, int N) {
    int idx = blockIdx.x * blockDim.x + threadIdx.x;
    if (idx >= outC * M) return;
    if (dsth) {
        int o = idx % outC;
        float2 st = stats[o];
        float v = (Y[idx] - st.x) * st.y * gamma[o] + beta[o];
        dsth[idx] = (_Float16)fmaxf(v, 0.f);
    } else {
        int n = idx % N;
        int o = (idx / N) % outC;
        int b = idx / (N * outC);
        float2 st = stats[o];
        float y = Y[(size_t)(b * N + n) * outC + o];
        float v = (y - st.x) * st.y * gamma[o] + beta[o];
        dstf[idx] = fmaxf(v, 0.f);
    }
}

// ---------------------------------------------------------------------------
// Host orchestration
// ---------------------------------------------------------------------------
static void run_layer(const float* Wf, const float* bf, const float* gf,
                      const float* bef, const _Float16* Xin, _Float16* Wh,
                      float* Y, float2* stats, _Float16* dsth, float* dstf,
                      int K, int outC, int M, int N, hipStream_t stream) {
    int nw = outC * K;
    f32_to_f16_kernel<<<(nw + 255) / 256, 256, 0, stream>>>(Wf, Wh, nw);
    int tiles = (M / 64) * (outC / 32);          // always divisible by 8 here
    gemm_wmma_kernel<<<tiles / 8, 256, 0, stream>>>(Wh, Xin, bf, Y, K, M, outC);
    stats_kernel<<<outC, 256, 0, stream>>>(Y, stats, M, outC);
    int tot = outC * M;
    bnrelu_kernel<<<(tot + 255) / 256, 256, 0, stream>>>(Y, gf, bef, stats,
                                                         dsth, dstf, outC, M, N);
}

extern "C" void kernel_launch(void* const* d_in, const int* in_sizes, int n_in,
                              void* d_out, int out_size, void* d_ws, size_t ws_size,
                              hipStream_t stream) {
    (void)in_sizes; (void)n_in; (void)out_size; (void)ws_size;
    const int B = 16;
    const int N0 = 4096, N1 = 1024, N2 = 256, N3 = 64, N4 = 16;

    const float* l0_xyz    = (const float*)d_in[0];
    const float* l1_xyz    = (const float*)d_in[2];
    const float* l1_points = (const float*)d_in[3];
    const float* l2_xyz    = (const float*)d_in[4];
    const float* l2_points = (const float*)d_in[5];
    const float* l3_xyz    = (const float*)d_in[6];
    const float* l3_points = (const float*)d_in[7];
    const float* l4_xyz    = (const float*)d_in[8];
    const float* l4_points = (const float*)d_in[9];
    // params: fp4 @10..17, fp3 @18..25, fp2 @26..33, fp1 @34..45
    const float* const* P = (const float* const*)d_in;

    char* ws = (char*)d_ws;
    _Float16* XA  = (_Float16*)(ws);                    // 20 MB  (X ping)
    _Float16* XB  = (_Float16*)(ws + (20ull << 20));    // 20 MB  (X pong)
    float*    Y   = (float*)   (ws + (40ull << 20));    // 36 MB  (GEMM out)
    _Float16* WH  = (_Float16*)(ws + (76ull << 20));    // 1 MB   (f16 weights)
    float2*   ST  = (float2*)  (ws + (77ull << 20));    // stats
    _Float16* L3P = (_Float16*)(ws + (78ull << 20));    // [1024, 256]
    _Float16* L2P = (_Float16*)(ws + (79ull << 20));    // [4096, 256]
    _Float16* L1P = (_Float16*)(ws + (82ull << 20));    // [16384, 128]
    float*    OUT = (float*)d_out;

    // ---------------- stage fp4: (l3, l4) -> l3p [B*64, 256] ----------------
    {
        int N = N3, S = N4, C1 = 256, C2 = 512, M = B * N, Kt = C1 + C2;
        copy_p1_kernel<<<(B * N * C1 + 255) / 256, 256, 0, stream>>>(l3_points, XA, B, C1, N, Kt);
        interp_kernel<<<(B * N + 255) / 256, 256, 0, stream>>>(
            l3_xyz, l4_xyz, l4_points, (const _Float16*)nullptr, XA, B, N, S, C2, C1, Kt);
        run_layer(P[10], P[11], P[12], P[13], XA, WH, Y, ST, XB, nullptr, 768, 256, M, N, stream);
        run_layer(P[14], P[15], P[16], P[17], XB, WH, Y, ST, L3P, nullptr, 256, 256, M, N, stream);
    }
    // ---------------- stage fp3: (l2, l3) -> l2p [B*256, 256] ---------------
    {
        int N = N2, S = N3, C1 = 128, C2 = 256, M = B * N, Kt = C1 + C2;
        copy_p1_kernel<<<(B * N * C1 + 255) / 256, 256, 0, stream>>>(l2_points, XA, B, C1, N, Kt);
        interp_kernel<<<(B * N + 255) / 256, 256, 0, stream>>>(
            l2_xyz, l3_xyz, (const float*)nullptr, L3P, XA, B, N, S, C2, C1, Kt);
        run_layer(P[18], P[19], P[20], P[21], XA, WH, Y, ST, XB, nullptr, 384, 256, M, N, stream);
        run_layer(P[22], P[23], P[24], P[25], XB, WH, Y, ST, L2P, nullptr, 256, 256, M, N, stream);
    }
    // ---------------- stage fp2: (l1, l2) -> l1p [B*1024, 128] --------------
    {
        int N = N1, S = N2, C1 = 64, C2 = 256, M = B * N, Kt = C1 + C2;
        copy_p1_kernel<<<(B * N * C1 + 255) / 256, 256, 0, stream>>>(l1_points, XA, B, C1, N, Kt);
        interp_kernel<<<(B * N + 255) / 256, 256, 0, stream>>>(
            l1_xyz, l2_xyz, (const float*)nullptr, L2P, XA, B, N, S, C2, C1, Kt);
        run_layer(P[26], P[27], P[28], P[29], XA, WH, Y, ST, XB, nullptr, 320, 256, M, N, stream);
        run_layer(P[30], P[31], P[32], P[33], XB, WH, Y, ST, L1P, nullptr, 256, 128, M, N, stream);
    }
    // ---------------- stage fp1: (l0, l1) -> out [B,128,4096] f32 -----------
    {
        int N = N0, S = N1, C2 = 128, M = B * N;
        interp_kernel<<<(B * N + 255) / 256, 256, 0, stream>>>(
            l0_xyz, l1_xyz, (const float*)nullptr, L1P, XA, B, N, S, C2, 0, C2);
        run_layer(P[34], P[35], P[36], P[37], XA, WH, Y, ST, XB, nullptr, 128, 128, M, N, stream);
        run_layer(P[38], P[39], P[40], P[41], XB, WH, Y, ST, XA, nullptr, 128, 128, M, N, stream);
        run_layer(P[42], P[43], P[44], P[45], XA, WH, Y, ST, (_Float16*)nullptr, OUT, 128, 128, M, N, stream);
    }
}
